// NODEModel_64209761075217
// MI455X (gfx1250) — matrix operand order repaired and checked by reading, not compiled, not run on hardware
//
#include <hip/hip_runtime.h>
#include <hip/hip_bf16.h>

typedef __attribute__((ext_vector_type(16))) __bf16 v16bf;
typedef __attribute__((ext_vector_type(8)))  float  v8f;

#define HSTR 136   // LDS row stride (bf16 elems) for activations: 272B -> bank step 4
#define WSTR 136   // LDS row stride for transposed weights

union Frag { uint4 u[2]; v16bf v; };  // 32 bytes = 16 bf16 = 8 VGPRs

__device__ __forceinline__ unsigned short f2bf(float f) {
    union { float f; unsigned u; } x; x.f = f;
    unsigned r = x.u + 0x7FFFu + ((x.u >> 16) & 1u);   // round-to-nearest-even
    return (unsigned short)(r >> 16);
}
__device__ __forceinline__ float bf2f(unsigned short h) {
    union { unsigned u; float f; } x; x.u = ((unsigned)h) << 16;
    return x.f;
}

// Branch-free tanh: 1 - 2/(exp2(2*log2(e)*x) + 1).
// 5 VALU ops (v_mul, v_exp_f32, v_add, v_rcp_f32, v_fma); exact +-1 saturation
// at large |x| (exp2 -> inf/0). Error ~1e-4, far below bf16 quantization.
__device__ __forceinline__ float fast_tanh(float x) {
    float t = __builtin_amdgcn_exp2f(x * 2.8853900817779268f);
    return 1.0f - 2.0f * __builtin_amdgcn_rcpf(t + 1.0f);
}

// ---------------------------------------------------------------------------
// Kernel 1: features. icIn[N][32] bf16 (K padded), dpIn[3N][32] bf16, a, 1/(1+r)
// ---------------------------------------------------------------------------
__global__ void featurize_kernel(const float* __restrict__ tx,
                                 unsigned short* __restrict__ icIn,
                                 unsigned short* __restrict__ dpIn,
                                 float* __restrict__ aArr,
                                 float* __restrict__ rinvArr,
                                 int n)
{
    int i = blockIdx.x * blockDim.x + threadIdx.x;
    if (i >= n) return;

    float tf = tx[4 * i + 0];
    float x  = tx[4 * i + 1];
    float y  = tx[4 * i + 2];
    float z  = tx[4 * i + 3];

    float r    = sqrtf(x * x + y * y + z * z);
    float rs   = fmaxf(r, 1e-8f);
    float ux   = x / rs, uy = y / rs, uz = z / rs;
    float rinv = 1.0f / (1.0f + r);
    float a    = tf * 0.5f;             // (tf - T0)*0.5 with T0 = 0

    aArr[i]    = a;
    rinvArr[i] = rinv;

    unsigned short fr  = f2bf(r),  fux = f2bf(ux), fuy = f2bf(uy);
    unsigned short fuz = f2bf(uz), fri = f2bf(rinv);

    // ic input row: [r, ux, uy, uz, rinv, 0...]
    {
        unsigned row[16];
        #pragma unroll
        for (int q = 0; q < 16; ++q) row[q] = 0u;
        row[0] = (unsigned)fr  | ((unsigned)fux << 16);
        row[1] = (unsigned)fuy | ((unsigned)fuz << 16);
        row[2] = (unsigned)fri;
        uint4* dst = (uint4*)(icIn + (size_t)i * 32);
        const uint4* src = (const uint4*)row;
        dst[0] = src[0]; dst[1] = src[1]; dst[2] = src[2]; dst[3] = src[3];
    }

    // dp input rows: [ts_j, r, ux, uy, uz, rinv, 0...], ts_j = a*(node_j + 1)
    const float nodes[3] = { -0.77459666924f, 0.0f, 0.77459666924f };
    #pragma unroll
    for (int j = 0; j < 3; ++j) {
        float tsj = a * nodes[j] + a;
        unsigned row[16];
        #pragma unroll
        for (int q = 0; q < 16; ++q) row[q] = 0u;
        row[0] = (unsigned)f2bf(tsj) | ((unsigned)fr  << 16);
        row[1] = (unsigned)fux       | ((unsigned)fuy << 16);
        row[2] = (unsigned)fuz       | ((unsigned)fri << 16);
        uint4* dst = (uint4*)(dpIn + ((size_t)i * 3 + j) * 32);
        const uint4* src = (const uint4*)row;
        dst[0] = src[0]; dst[1] = src[1]; dst[2] = src[2]; dst[3] = src[3];
    }
}

// ---------------------------------------------------------------------------
// Kernel 2: fused MLP.  in: [M,32] bf16 (K-padded);  Win: [din,128] f32;
// Wh: [3,128,128] f32; Wout: [128] f32.  128 threads = 4 waves = 64 rows/block.
// ---------------------------------------------------------------------------
__global__ void __launch_bounds__(128)
mlp_fused_kernel(const unsigned short* __restrict__ in,
                 const float* __restrict__ Win, int din,
                 const float* __restrict__ bin,
                 const float* __restrict__ Wh,
                 const float* __restrict__ bh,
                 const float* __restrict__ Wout,
                 const float* __restrict__ bout,
                 float* __restrict__ out)
{
    __shared__ unsigned short H [64  * HSTR];  // activations, bf16
    __shared__ unsigned short Wt[128 * WSTR];  // transposed weights Wt[n][k], bf16

    const int t    = threadIdx.x;
    const int lane = t & 31;
    const int wave = t >> 5;
    const int m0   = wave * 16;                 // this wave's M-tile
    const long rowBase = (long)blockIdx.x * 64;

    const int mrow = lane & 15;                 // A-frag / row within tile
    const int kA   = (lane >> 4) * 8;           // A-frag K sub-base (0 / 8)
    const int kB   = (lane >> 4) * 8;           // B-frag K sub-base (0 / 8)
    const int ncol = lane & 15;                 // B/C-frag column within tile

    // ---- stage layer-0 weights: Wt[n][k] = (k < din) ? Win[k*128+n] : 0 ----
    for (int e = t; e < 32 * 128; e += 128) {
        int k = e >> 7, n = e & 127;            // e == k*128 + n == flat Win idx
        float v = (k < din) ? Win[e] : 0.0f;
        Wt[n * WSTR + k] = f2bf(v);
    }
    // prefetch first hidden layer's weights into cache while we compute
    #pragma unroll
    for (int j = 0; j < 4; ++j)
        __builtin_prefetch(Wh + t * 128 + j * 32, 0, 1);
    __syncthreads();

    // ---- layer 0: H = tanh(in @ Win + bin), K = 32 (one K-step) ----
    {
        Frag a;
        const unsigned short* rp = in + (size_t)(rowBase + m0 + mrow) * 32;
        a.u[0] = *(const uint4*)(rp + kA);
        a.u[1] = *(const uint4*)(rp + 16 + kA);

        #pragma unroll
        for (int nt = 0; nt < 8; ++nt) {
            v8f acc = {};
            Frag b;
            const unsigned short* wp = &Wt[(nt * 16 + ncol) * WSTR + kB];
            b.u[0] = *(const uint4*)(wp);
            b.u[1] = *(const uint4*)(wp + 16);
            acc = __builtin_amdgcn_wmma_f32_16x16x32_bf16(
                      false, a.v, false, b.v, (short)0, acc, false, false);
            float bias = bin[nt * 16 + ncol];
            #pragma unroll
            for (int r = 0; r < 8; ++r) {
                float h = fast_tanh(acc[r] + bias);
                H[(m0 + (lane >> 4) * 8 + r) * HSTR + nt * 16 + ncol] = f2bf(h);
            }
        }
    }
    __syncthreads();

    // ---- hidden layers l = 0..2: H = tanh(H @ Wh[l] + bh[l]), K = 128 ----
    for (int l = 0; l < 3; ++l) {
        const float* Wl = Wh + (size_t)l * 128 * 128;
        for (int e = t; e < 128 * 128; e += 128) {
            int k = e >> 7, n = e & 127;        // e == flat Wl idx (coalesced)
            Wt[n * WSTR + k] = f2bf(Wl[e]);
        }
        // prefetch next layer's weights while this layer computes
        if (l < 2) {
            const float* Wnext = Wl + 128 * 128;
            #pragma unroll
            for (int j = 0; j < 4; ++j)
                __builtin_prefetch(Wnext + t * 128 + j * 32, 0, 1);
        }
        __syncthreads();

        // load all 4 A K-step fragments for this wave's rows (before any store)
        Frag af[4];
        const unsigned short* hp = &H[(m0 + mrow) * HSTR];
        #pragma unroll
        for (int kt = 0; kt < 4; ++kt) {
            af[kt].u[0] = *(const uint4*)(hp + kt * 32 + kA);
            af[kt].u[1] = *(const uint4*)(hp + kt * 32 + 16 + kA);
        }

        #pragma unroll
        for (int nt = 0; nt < 8; ++nt) {
            v8f acc = {};
            #pragma unroll
            for (int kt = 0; kt < 4; ++kt) {
                Frag b;
                const unsigned short* wp =
                    &Wt[(nt * 16 + ncol) * WSTR + kt * 32 + kB];
                b.u[0] = *(const uint4*)(wp);
                b.u[1] = *(const uint4*)(wp + 16);
                acc = __builtin_amdgcn_wmma_f32_16x16x32_bf16(
                          false, af[kt].v, false, b.v, (short)0, acc, false, false);
            }
            float bias = bh[l * 128 + nt * 16 + ncol];
            #pragma unroll
            for (int r = 0; r < 8; ++r) {
                float h = fast_tanh(acc[r] + bias);
                H[(m0 + (lane >> 4) * 8 + r) * HSTR + nt * 16 + ncol] = f2bf(h);
            }
        }
        __syncthreads();
    }

    // ---- output layer: out[row] = H[row,:] . Wout + bout ----
    if (t < 64) {
        float s = 0.0f;
        const unsigned short* hp = &H[t * HSTR];
        #pragma unroll 4
        for (int k = 0; k < 128; ++k)
            s += bf2f(hp[k]) * Wout[k];
        out[rowBase + t] = s + bout[0];
    }
}

// ---------------------------------------------------------------------------
// Kernel 3: combine quadrature + initial correction, scale by 1/(1+r)
// ---------------------------------------------------------------------------
__global__ void combine_kernel(const float* __restrict__ icOut,
                               const float* __restrict__ dpOut,
                               const float* __restrict__ aArr,
                               const float* __restrict__ rinvArr,
                               float* __restrict__ out, int n)
{
    int i = blockIdx.x * blockDim.x + threadIdx.x;
    if (i >= n) return;
    const float w0 = 5.0f / 9.0f, w1 = 8.0f / 9.0f, w2 = 5.0f / 9.0f;
    float s = w0 * dpOut[3 * i + 0] + w1 * dpOut[3 * i + 1] + w2 * dpOut[3 * i + 2];
    out[i] = (icOut[i] + aArr[i] * s) * rinvArr[i];
}

// ---------------------------------------------------------------------------
extern "C" void kernel_launch(void* const* d_in, const int* in_sizes, int n_in,
                              void* d_out, int out_size, void* d_ws, size_t ws_size,
                              hipStream_t stream)
{
    const float* tx      = (const float*)d_in[0];
    const float* dp_Win  = (const float*)d_in[1];
    const float* dp_bin  = (const float*)d_in[2];
    const float* dp_Wh   = (const float*)d_in[3];
    const float* dp_bh   = (const float*)d_in[4];
    const float* dp_Wout = (const float*)d_in[5];
    const float* dp_bout = (const float*)d_in[6];
    const float* ic_Win  = (const float*)d_in[7];
    const float* ic_bin  = (const float*)d_in[8];
    const float* ic_Wh   = (const float*)d_in[9];
    const float* ic_bh   = (const float*)d_in[10];
    const float* ic_Wout = (const float*)d_in[11];
    const float* ic_bout = (const float*)d_in[12];

    const int N = in_sizes[0] / 4;

    char* ws = (char*)d_ws;
    size_t off = 0;
    unsigned short* icIn = (unsigned short*)(ws + off); off += (size_t)N * 32 * 2;
    unsigned short* dpIn = (unsigned short*)(ws + off); off += (size_t)N * 3 * 32 * 2;
    float* icOut   = (float*)(ws + off); off += (size_t)N * 4;
    float* dpOut   = (float*)(ws + off); off += (size_t)N * 3 * 4;
    float* aArr    = (float*)(ws + off); off += (size_t)N * 4;
    float* rinvArr = (float*)(ws + off); off += (size_t)N * 4;

    featurize_kernel<<<(N + 255) / 256, 256, 0, stream>>>(
        tx, icIn, dpIn, aArr, rinvArr, N);

    // dp MLP: batch 3N, d_in = 6
    mlp_fused_kernel<<<(3 * N) / 64, 128, 0, stream>>>(
        dpIn, dp_Win, 6, dp_bin, dp_Wh, dp_bh, dp_Wout, dp_bout, dpOut);

    // ic MLP: batch N, d_in = 5
    mlp_fused_kernel<<<N / 64, 128, 0, stream>>>(
        icIn, ic_Win, 5, ic_bin, ic_Wh, ic_bh, ic_Wout, ic_bout, icOut);

    combine_kernel<<<(N + 255) / 256, 256, 0, stream>>>(
        icOut, dpOut, aArr, rinvArr, (float*)d_out, N);
}
